// GraphNorm_798863917425
// MI455X (gfx1250) — matrix operand order repaired and checked by reading, not compiled, not run on hardware
//
#include <hip/hip_runtime.h>
#include <hip/hip_bf16.h>

#define HID 256
#define TWOH 512
#define NSEG 4096
#define EPSV 1e-5f

typedef __attribute__((ext_vector_type(16))) __bf16 bf16x16;
typedef __attribute__((ext_vector_type(8)))  __bf16 bf16x8;
typedef __attribute__((ext_vector_type(8)))  float  f32x8;

// ---------------- zero workspace accumulators ----------------
__global__ void gn_zero(float* __restrict__ p, int n) {
    int i = blockIdx.x * blockDim.x + threadIdx.x;
    if (i < n) p[i] = 0.0f;
}

// ---------------- convert W_ada (f32 [512,256]) -> bf16 ----------------
__global__ void gn_wconv(const float* __restrict__ W, unsigned short* __restrict__ wb, int n) {
    int i = blockIdx.x * blockDim.x + threadIdx.x;
    if (i < n) ((__bf16*)wb)[i] = (__bf16)W[i];
}

// ---------------- segment stats: sum, sumsq, counts (batch sorted) ----------------
__global__ __launch_bounds__(256)
void gn_stats(const float* __restrict__ h, const int* __restrict__ batch,
              float* __restrict__ sums, float* __restrict__ sumsq,
              float* __restrict__ counts, int n) {
    __shared__ int sb[128];
    const int t  = threadIdx.x;          // channel 0..255
    const int i0 = blockIdx.x * 128;
    const int cn = (n - i0 < 128) ? (n - i0) : 128;
    if (t < cn) sb[t] = batch[i0 + t];
    __syncthreads();
    if (cn <= 0) return;

    float s1 = 0.0f, s2 = 0.0f;
    int   cur = sb[0];
    int   run = 0;
    for (int j = 0; j < cn; ++j) {
        const int b = sb[j];             // uniform across block
        if (b != cur) {
            atomicAdd(&sums [cur * HID + t], s1);
            atomicAdd(&sumsq[cur * HID + t], s2);
            if (t == 0) atomicAdd(&counts[cur], (float)run);
            s1 = 0.0f; s2 = 0.0f; run = 0; cur = b;
        }
        const float v = h[(size_t)(i0 + j) * HID + t];
        s1 += v; s2 += v * v; ++run;
    }
    atomicAdd(&sums [cur * HID + t], s1);
    atomicAdd(&sumsq[cur * HID + t], s2);
    if (t == 0) atomicAdd(&counts[cur], (float)run);
}

// ---------------- finalize: mu, rstd per (segment, channel) ----------------
// var(h - a*mu) = E[h^2] - (2a - a^2) * mu^2
__global__ __launch_bounds__(256)
void gn_finalize(float* __restrict__ sums, float* __restrict__ sumsq,
                 const float* __restrict__ counts, const float* __restrict__ alpha) {
    const int idx = blockIdx.x * blockDim.x + threadIdx.x;  // NSEG*HID threads
    const int seg = idx >> 8;
    const int c   = idx & 255;
    float cnt = counts[seg];
    cnt = (cnt < 1.0f) ? 1.0f : cnt;
    const float inv = 1.0f / cnt;
    const float mu  = sums[idx] * inv;
    const float e2  = sumsq[idx] * inv;
    const float a   = alpha[c];
    const float var = e2 - (2.0f * a - a * a) * mu * mu;
    sums[idx]  = mu;                     // reuse in place
    sumsq[idx] = rsqrtf(var + EPSV);     // reuse in place
}

// ---------------- fused GEMM (bf16 WMMA) + GraphNorm epilogue ----------------
// Block: 256 threads = 8 waves; block tile = 32 nodes x 512 output cols.
// Wave w: M-tile = (w>>2)*16, scale cols [c0, c0+64) with c0=(w&3)*64,
//         and matching shift cols [c0+256, c0+320).
__global__ __launch_bounds__(256)
void gn_main(const float* __restrict__ h, const float* __restrict__ y,
             const int* __restrict__ batch, const float* __restrict__ alpha,
             const float* __restrict__ b_ada, const unsigned short* __restrict__ wbf_u,
             const float* __restrict__ mu, const float* __restrict__ rstd,
             float* __restrict__ out, int n) {
    const __bf16* wbf = (const __bf16*)wbf_u;
    __shared__ __bf16 lds_y[32 * HID];   // 16 KB: y tile as bf16

    const int t  = threadIdx.x;
    const int m0 = blockIdx.x * 32;

    // Stage y[m0 .. m0+32) into LDS as bf16 (thread t owns channel t).
    #pragma unroll
    for (int j = 0; j < 32; ++j) {
        const int node = m0 + j;
        const float v  = (node < n) ? y[(size_t)node * HID + t] : 0.0f;
        lds_y[j * HID + t] = (__bf16)v;
    }
    __syncthreads();

    const int wave  = t >> 5;
    const int lane  = t & 31;
    const int mbase = (wave >> 2) * 16;          // 0 or 16
    const int c0    = (wave & 3) * 64;           // scale column base
    const int lcol  = lane & 15;
    const int koff  = (lane < 16) ? 0 : 16;      // A-matrix 16-bit lane layout
    const int bkoff = (lane < 16) ? 0 : 16;      // B-matrix 16-bit lane layout

    f32x8 acc[8] = {};                            // 4 scale tiles + 4 shift tiles

    #pragma unroll
    for (int kc = 0; kc < 8; ++kc) {
        const int k0 = kc * 32;
        // A fragment (16x32 bf16) from LDS, contiguous k order (same as B):
        // lane<16: elems0-15=K k0..k0+15 ; lane>=16: +16
        const int arow = mbase + lcol;
        const bf16x8 alo = *(const bf16x8*)&lds_y[arow * HID + k0 + koff];
        const bf16x8 ahi = *(const bf16x8*)&lds_y[arow * HID + k0 + koff + 8];
        const bf16x16 afrag = __builtin_shufflevector(
            alo, ahi, 0, 1, 2, 3, 4, 5, 6, 7, 8, 9, 10, 11, 12, 13, 14, 15);

        #pragma unroll
        for (int tt = 0; tt < 8; ++tt) {
            // B column = output index; column n of B == row n of W (contiguous K)
            const int col = c0 + (tt & 3) * 16 + ((tt >= 4) ? HID : 0) + lcol;
            const bf16x16 bfrag = *(const bf16x16*)&wbf[(size_t)col * HID + k0 + bkoff];
            acc[tt] = __builtin_amdgcn_wmma_f32_16x16x32_bf16(
                false, afrag, false, bfrag, (short)0, acc[tt], false, false);
        }
    }

    // Epilogue: C/D layout — VGPR r, lane L: row = r + 8*(L>>4), col = tile + (L&15)
    #pragma unroll
    for (int r = 0; r < 8; ++r) {
        const int node = m0 + mbase + r + 8 * (lane >> 4);
        if (node >= n) continue;
        const int    seg   = batch[node];
        const size_t hbase = (size_t)node * HID;
        const size_t sbase = (size_t)seg * HID;
        #pragma unroll
        for (int tt = 0; tt < 4; ++tt) {
            const int   c     = c0 + tt * 16 + lcol;
            const float scale = acc[tt][r]     + b_ada[c];
            const float shift = acc[tt + 4][r] + b_ada[c + HID];
            const float hv    = h[hbase + c];
            const float m     = mu[sbase + c];
            const float rs    = rstd[sbase + c];
            const float nh    = (hv - alpha[c] * m) * rs;
            out[hbase + c] = shift + nh * (scale + 1.0f);
        }
    }
}

extern "C" void kernel_launch(void* const* d_in, const int* in_sizes, int n_in,
                              void* d_out, int out_size, void* d_ws, size_t ws_size,
                              hipStream_t stream) {
    const float* h     = (const float*)d_in[0];
    const float* y     = (const float*)d_in[1];
    const int*   batch = (const int*)d_in[2];
    const float* alpha = (const float*)d_in[3];
    const float* W     = (const float*)d_in[4];
    const float* b_ada = (const float*)d_in[5];
    float*       out   = (float*)d_out;
    const int n = in_sizes[2];               // number of nodes

    // Workspace layout
    char* ws = (char*)d_ws;
    float* sums   = (float*)(ws);                         // 4096*256 f32 = 4 MB (-> mu)
    float* sumsq  = (float*)(ws + 4194304);               // 4 MB (-> rstd)
    float* counts = (float*)(ws + 8388608);               // 16 KB
    unsigned short* wbf = (unsigned short*)(ws + 8404992);// 512*256 bf16 = 256 KB

    // 1) zero accumulators (sums, sumsq, counts are contiguous)
    const int nz = NSEG * HID * 2 + NSEG;
    gn_zero<<<(nz + 255) / 256, 256, 0, stream>>>(sums, nz);

    // 2) W -> bf16 (once; 256 KB stays L2-resident across the big kernel)
    const int nw = TWOH * HID;
    gn_wconv<<<(nw + 255) / 256, 256, 0, stream>>>(W, wbf, nw);

    // 3) segment stats (one pass over h, run-length + atomics)
    gn_stats<<<(n + 127) / 128, 256, 0, stream>>>(h, batch, sums, sumsq, counts, n);

    // 4) finalize mu / rstd
    gn_finalize<<<(NSEG * HID) / 256, 256, 0, stream>>>(sums, sumsq, counts, alpha);

    // 5) fused WMMA GEMM + GraphNorm epilogue
    gn_main<<<(n + 31) / 32, 256, 0, stream>>>(h, y, batch, alpha, b_ada, wbf,
                                               sums, sumsq, out, n);
}
